// GNPP_42253888258134
// MI455X (gfx1250) — matrix operand order repaired and checked by reading, not compiled, not run on hardware
//
#include <hip/hip_runtime.h>

// ---------------------------------------------------------------------------
// Fused harmonic-attention for MI455X (gfx1250, wave32, WMMA).
//
//  N=8192 queries, L=4096 sources, E=128, H=4 heads.
//  out = [ atten_outputs (N x H*E) | attn (H x N x L) ]  (fp32)
//
//  Precompute (in d_ws):
//    phi_t  [H][N][E]  f16 row-major        (A operand, GEMM1)        8 MB
//    phi_ts [H][L][E]  f16 row-major        (B operand, GEMM1)        4 MB
//    pvT    [H][E][L]  f16 = (phi_ts @ V)^T (B operand, GEMM2)        4 MB
//    packT  [N/16][2][L] u8: bit r = mask[qt*16 + hf*8 + r][l]        4 MB
//
//  Main kernel: one wave32 per (head, 16-query tile). |S| <= 2 by construction
//  so softmax needs no max-subtraction: pass 1 = row sums of exp(S) (masked),
//  pass 2 = recompute S, stream normalized attn with NT stores, accumulate
//  out with WMMA on unnormalized P=exp(S), final row scale by 1/rowsum.
// ---------------------------------------------------------------------------

typedef _Float16 f16;
typedef __attribute__((ext_vector_type(8)))  _Float16 v8h;
typedef __attribute__((ext_vector_type(16))) _Float16 v16h;
typedef __attribute__((ext_vector_type(8)))  float    v8f;

namespace {
constexpr int kN  = 8192;
constexpr int kL  = 4096;
constexpr int kE  = 128;
constexpr int kH  = 4;
constexpr int kEh = kE / 2;
}

__device__ __forceinline__ v16h cat16(v8h lo, v8h hi) {
  return __builtin_shufflevector(lo, hi, 0, 1, 2, 3, 4, 5, 6, 7,
                                 8, 9, 10, 11, 12, 13, 14, 15);
}

// phi[h][t][e] = 0.125 * {cos,sin}(ts[t] * freq[h][j]),  e = j / 64+j
__global__ void phi_kernel(const float* __restrict__ ts,
                           const float* __restrict__ freq,
                           f16* __restrict__ phi, int T) {
  int idx = blockIdx.x * blockDim.x + threadIdx.x;
  if (idx >= kH * T * kEh) return;
  int j = idx % kEh;
  int r = idx / kEh;
  int t = r % T;
  int h = r / T;
  float m = ts[t] * freq[h * kEh + j];
  float s, c;
  __sincosf(m, &s, &c);
  size_t base = ((size_t)h * T + t) * kE;
  phi[base + j]       = (f16)(0.125f * c);
  phi[base + kEh + j] = (f16)(0.125f * s);
}

// packT[qt][hf][l] : bit r = mask[qt*16 + hf*8 + r][l]
__global__ void pack_kernel(const unsigned char* __restrict__ mask,
                            unsigned char* __restrict__ packT) {
  int idx = blockIdx.x * blockDim.x + threadIdx.x;  // (N/16)*2*L
  if (idx >= (kN / 16) * 2 * kL) return;
  int l  = idx % kL;
  int r  = idx / kL;
  int hf = r & 1;
  int qt = r >> 1;
  const unsigned char* mrow = mask + (size_t)(qt * 16 + hf * 8) * kL + l;
  unsigned int b = 0;
#pragma unroll
  for (int i = 0; i < 8; ++i)
    b |= (mrow[(size_t)i * kL] ? 1u : 0u) << i;
  packT[idx] = (unsigned char)b;
}

// pvT[h][f][l] = sum_e phi_ts[h][l][e] * V[e][f]   (stored transposed, f16)
// One wave owns 16 features (wave-uniform -> V reads are scalar broadcasts),
// one lane owns one l (coalesced phi reads / pvT writes).
__launch_bounds__(256)
__global__ void pv_kernel(const f16* __restrict__ phi_ts,
                          const float* __restrict__ V,
                          f16* __restrict__ pvT) {
  const int lane = threadIdx.x & 31;
  const int wv   = threadIdx.x >> 5;            // 8 waves -> f0 = 0..112
  const int blk  = blockIdx.x;                  // kH * (kL/32)
  const int h    = blk / (kL / 32);
  const int l    = (blk % (kL / 32)) * 32 + lane;
  const int f0   = wv * 16;
  const f16* prow = phi_ts + ((size_t)h * kL + l) * kE;
  float acc[16];
#pragma unroll
  for (int i = 0; i < 16; ++i) acc[i] = 0.f;
  for (int e = 0; e < kE; ++e) {
    float p = (float)prow[e];
#pragma unroll
    for (int i = 0; i < 16; ++i) acc[i] += p * V[e * kE + f0 + i];
  }
#pragma unroll
  for (int i = 0; i < 16; ++i)
    pvT[((size_t)h * kE + f0 + i) * kL + l] = (f16)acc[i];
}

__launch_bounds__(256)
__global__ void attn_kernel(const f16* __restrict__ phi_t,
                            const f16* __restrict__ phi_ts,
                            const f16* __restrict__ pvT,
                            const unsigned char* __restrict__ packT,
                            float* __restrict__ out) {
  __shared__ __align__(16) f16 lds[8][16][32];  // per-wave 16x32 P staging

  const int lane  = threadIdx.x & 31;
  const int widx  = threadIdx.x >> 5;
  const int wid   = blockIdx.x * 8 + widx;   // 2048 wave tasks
  const int h     = wid >> 9;                // / (N/16)
  const int qt    = wid & 511;
  const int qbase = qt * 16;

  const int nlo = lane & 15;       // B/C/D column within tile; A row
  const int hf  = lane >> 4;       // lane half
  const int abk = hf ? 8 : 0;      // A-fragment base_k (16-bit A 16x32 layout)
  const int bbk = hf ? 8 : 0;      // B-fragment base_k (32x16 layout)

  // Resident A fragments: this wave's 16x128 phi_t tile (4 chunks of K=32).
  v16h afrag[4];
  {
    const f16* arow = phi_t + ((size_t)h * kN + qbase + nlo) * kE;
#pragma unroll
    for (int kc = 0; kc < 4; ++kc) {
      v8h lo = *(const v8h*)(arow + kc * 32 + abk);
      v8h hi = *(const v8h*)(arow + kc * 32 + 16 + abk);
      afrag[kc] = cat16(lo, hi);
    }
  }

  const f16* bbase = phi_ts + ((size_t)h * kL + nlo) * kE;
  // Per-lane packed-mask pointer: mlane[l] has the 8 row-bits for column l.
  const unsigned char* mlane = packT + ((size_t)qt * 2 + hf) * kL + nlo;

  // ---------------- Pass 1: rowsum of exp(S) over unmasked columns ---------
  float rsum[8];
#pragma unroll
  for (int r = 0; r < 8; ++r) rsum[r] = 0.f;

  for (int l0 = 0; l0 < kL; l0 += 16) {
    const f16* brow = bbase + (size_t)l0 * kE;
    v8f s = {};
#pragma unroll
    for (int kc = 0; kc < 4; ++kc) {
      v8h lo = *(const v8h*)(brow + kc * 32 + bbk);
      v8h hi = *(const v8h*)(brow + kc * 32 + bbk + 16);
      s = __builtin_amdgcn_wmma_f32_16x16x32_f16(false, afrag[kc], false,
                                                 cat16(lo, hi), (short)0, s,
                                                 false, false);
    }
    const unsigned int mb = mlane[l0];
#pragma unroll
    for (int r = 0; r < 8; ++r) {
      if (!(mb & (1u << r))) rsum[r] += __expf(s[r]);
    }
  }
#pragma unroll
  for (int r = 0; r < 8; ++r) {
#pragma unroll
    for (int off = 1; off < 16; off <<= 1)
      rsum[r] += __shfl_xor(rsum[r], off, 32);   // reduce within 16-lane half
  }
  float inv[8];
#pragma unroll
  for (int r = 0; r < 8; ++r) inv[r] = rsum[r] > 0.f ? 1.f / rsum[r] : 0.f;

  // ---------------- Pass 2: attn write + out accumulation ------------------
  float* attn = out + (size_t)kN * (kH * kE) + (size_t)h * kN * kL;
  v8f oacc[8];
#pragma unroll
  for (int ft = 0; ft < 8; ++ft) oacc[ft] = v8f{};

  for (int l0 = 0; l0 < kL; l0 += 32) {
#pragma unroll
    for (int j = 0; j < 2; ++j) {
      const int lt = l0 + j * 16;
      const f16* brow = bbase + (size_t)lt * kE;
      v8f s = {};
#pragma unroll
      for (int kc = 0; kc < 4; ++kc) {
        v8h lo = *(const v8h*)(brow + kc * 32 + bbk);
        v8h hi = *(const v8h*)(brow + kc * 32 + bbk + 16);
        s = __builtin_amdgcn_wmma_f32_16x16x32_f16(false, afrag[kc], false,
                                                   cat16(lo, hi), (short)0, s,
                                                   false, false);
      }
      const unsigned int mb = mlane[lt];
      const int lc = lt + nlo;
#pragma unroll
      for (int r = 0; r < 8; ++r) {
        const int m = r + hf * 8;
        const float p = (mb & (1u << r)) ? 0.f : __expf(s[r]);
        // normalized attn, streaming (non-temporal) store
        __builtin_nontemporal_store(p * inv[r],
                                    &attn[(size_t)(qbase + m) * kL + lc]);
        lds[widx][m][j * 16 + nlo] = (f16)p;   // unnormalized P for GEMM2
      }
    }
    asm volatile("s_wait_dscnt 0" ::: "memory");
    // Reload P tile as an A fragment (16x32, 16-bit A layout).
    v8h plo = *(const v8h*)(&lds[widx][nlo][abk]);
    v8h phi = *(const v8h*)(&lds[widx][nlo][16 + abk]);
    const v16h pfrag = cat16(plo, phi);
#pragma unroll
    for (int ft = 0; ft < 8; ++ft) {
      const f16* pvrow = pvT + ((size_t)h * kE + ft * 16 + nlo) * kL + l0;
      v8h lo = *(const v8h*)(pvrow + bbk);
      v8h hi = *(const v8h*)(pvrow + bbk + 16);
      oacc[ft] = __builtin_amdgcn_wmma_f32_16x16x32_f16(false, pfrag, false,
                                                        cat16(lo, hi), (short)0,
                                                        oacc[ft], false, false);
    }
  }

  // atten_outputs[n][h*E + f] = out_row(n) scaled by 1/rowsum
#pragma unroll
  for (int ft = 0; ft < 8; ++ft) {
#pragma unroll
    for (int r = 0; r < 8; ++r) {
      const int m = qbase + r + hf * 8;
      out[(size_t)m * (kH * kE) + h * kE + ft * 16 + nlo] = oacc[ft][r] * inv[r];
    }
  }
}

extern "C" void kernel_launch(void* const* d_in, const int* in_sizes, int n_in,
                              void* d_out, int out_size, void* d_ws, size_t ws_size,
                              hipStream_t stream) {
  const float* t            = (const float*)d_in[0];        // [N]
  const float* ets          = (const float*)d_in[1];        // [L]
  const unsigned char* mask = (const unsigned char*)d_in[2];// [N][L] bool
  const float* freq         = (const float*)d_in[3];        // [H][E/2]
  const float* V            = (const float*)d_in[4];        // [E][E]
  float* out                = (float*)d_out;

  char* ws = (char*)d_ws;
  f16* phi_t            = (f16*)ws;                               // 8 MB
  f16* phi_ts           = (f16*)(ws + (size_t)8  * 1024 * 1024);  // 4 MB
  f16* pvT              = (f16*)(ws + (size_t)12 * 1024 * 1024);  // 4 MB
  unsigned char* packT  = (unsigned char*)(ws + (size_t)16 * 1024 * 1024); // 4 MB

  {
    int tot = kH * kN * kEh;
    phi_kernel<<<(tot + 255) / 256, 256, 0, stream>>>(t, freq, phi_t, kN);
  }
  {
    int tot = kH * kL * kEh;
    phi_kernel<<<(tot + 255) / 256, 256, 0, stream>>>(ets, freq, phi_ts, kL);
  }
  {
    int tot = (kN / 16) * 2 * kL;
    pack_kernel<<<(tot + 255) / 256, 256, 0, stream>>>(mask, packT);
  }
  pv_kernel<<<kH * (kL / 32), 256, 0, stream>>>(phi_ts, V, pvT);
  attn_kernel<<<(kH * (kN / 16)) / 8, 256, 0, stream>>>(phi_t, phi_ts, pvT,
                                                        packT, out);
  (void)in_sizes; (void)n_in; (void)out_size; (void)ws_size;
}
